// GraphTransformer_54726473285924
// MI455X (gfx1250) — compile-verified, matrix operand order checked by reading, not run on hardware
//
#include <hip/hip_runtime.h>
#include <hip/hip_bf16.h>

// ---------------------------------------------------------------------------
// Graph Transformer layer for MI455X (gfx1250, wave32, WMMA).
// GEMMs: bf16 WMMA (v_wmma_f32_16x16x32_bf16), fp32 accumulate, software-
// pipelined K loop. Attention: CSR-by-dst + per-wave online softmax (no float
// atomics).
// ---------------------------------------------------------------------------

typedef __bf16 bf16_t;
typedef __attribute__((ext_vector_type(16))) __bf16 v16bf;
typedef __attribute__((ext_vector_type(8)))  float  v8f;

#define DIM    128
#define HEADS  8

// ---------------------------------------------------------------------------
// Pack a row-major fp32 weight [K][Ncols] into WMMA-lane-ordered bf16 frags.
// Fragment (nt, kt): 32 lanes x 16 bf16. Lane l<16: col nt*16+l,
// K = kt*32 + {0..7, 16..23}; lane l>=16: col nt*16+(l-16),
// K = kt*32 + {8..15, 24..31}  (16-bit B VGPR striping).
// ---------------------------------------------------------------------------
__global__ void pack_weights(const float* __restrict__ W, bf16_t* __restrict__ P,
                             int K, int Ncols) {
  int tid = blockIdx.x * blockDim.x + threadIdx.x;
  int ktiles = K >> 5;
  int ntiles = Ncols >> 4;
  int total = ntiles * ktiles * 32;
  if (tid >= total) return;
  int lane = tid & 31;
  int kt   = (tid >> 5) % ktiles;
  int nt   = (tid >> 5) / ktiles;
  int col  = nt * 16 + (lane & 15);
  int koff = (lane < 16) ? 0 : 8;
  int kb   = kt * 32;
  bf16_t* dst = P + (size_t)tid * 16;
#pragma unroll
  for (int j = 0; j < 8; ++j)
    dst[j] = (bf16_t)W[(size_t)(kb + koff + j) * Ncols + col];
#pragma unroll
  for (int j = 0; j < 8; ++j)
    dst[8 + j] = (bf16_t)W[(size_t)(kb + koff + 16 + j) * Ncols + col];
}

// ---------------------------------------------------------------------------
// WMMA GEMM helpers
// ---------------------------------------------------------------------------
__device__ __forceinline__ v16bf load_a_frag(const float* __restrict__ Arow,
                                             int kt, int lane) {
  const int k1 = kt * 32 + ((lane < 16) ? 0 : 8);
  const float4 f0 = *(const float4*)(Arow + k1);
  const float4 f1 = *(const float4*)(Arow + k1 + 4);
  const float4 f2 = *(const float4*)(Arow + k1 + 16);
  const float4 f3 = *(const float4*)(Arow + k1 + 20);
  v16bf a;
  a[0]  = (bf16_t)f0.x; a[1]  = (bf16_t)f0.y;
  a[2]  = (bf16_t)f0.z; a[3]  = (bf16_t)f0.w;
  a[4]  = (bf16_t)f1.x; a[5]  = (bf16_t)f1.y;
  a[6]  = (bf16_t)f1.z; a[7]  = (bf16_t)f1.w;
  a[8]  = (bf16_t)f2.x; a[9]  = (bf16_t)f2.y;
  a[10] = (bf16_t)f2.z; a[11] = (bf16_t)f2.w;
  a[12] = (bf16_t)f3.x; a[13] = (bf16_t)f3.y;
  a[14] = (bf16_t)f3.z; a[15] = (bf16_t)f3.w;
  return a;
}

__device__ __forceinline__ v16bf load_b_frag(const bf16_t* __restrict__ Bp,
                                             int fragIdx) {
  return *(const v16bf*)(Bp + (size_t)fragIdx * 16);
}

// ---------------------------------------------------------------------------
// C[:, coloff:+N] = act(A @ B + bias (+ resid));  A: [M][K] fp32 (lda == K).
// Block tile 64x128, 8 waves; wave: 16 rows x 64 cols = 4 WMMA accumulators.
// KTILES = K/32 (compile-time) -> fully unrolled, prefetch-1 pipelined.
// resid (if non-null) has leading dim 128.
// ---------------------------------------------------------------------------
template <int KTILES>
__global__ __launch_bounds__(256)
void gemm_wmma(const float* __restrict__ A, const bf16_t* __restrict__ Bp,
               const float* __restrict__ bias, const float* __restrict__ resid,
               float* __restrict__ C, int M, int ldc, int coloff,
               int relu_flag) {
  const int lane = threadIdx.x & 31;
  const int wave = threadIdx.x >> 5;
  const int wm = wave & 3;                 // 0..3 -> 16-row slabs
  const int wn = wave >> 2;                // 0..1 -> 64-col slabs
  const int rowbase = blockIdx.x * 64 + wm * 16;
  const int nbase   = blockIdx.y * 128 + wn * 64;

  v8f acc[4];
  v8f zero = {};
#pragma unroll
  for (int i = 0; i < 4; ++i) acc[i] = zero;

  int arow = rowbase + (lane & 15);
  if (arow >= M) arow = M - 1;
  const float* Arow = A + (size_t)arow * (KTILES * 32);
  const int ntb = nbase >> 4;

  // prologue: fragments for kt = 0
  v16bf af = load_a_frag(Arow, 0, lane);
  v16bf b0 = load_b_frag(Bp, ((ntb + 0) * KTILES + 0) * 32 + lane);
  v16bf b1 = load_b_frag(Bp, ((ntb + 1) * KTILES + 0) * 32 + lane);
  v16bf b2 = load_b_frag(Bp, ((ntb + 2) * KTILES + 0) * 32 + lane);
  v16bf b3 = load_b_frag(Bp, ((ntb + 3) * KTILES + 0) * 32 + lane);

#pragma unroll
  for (int kt = 0; kt < KTILES; ++kt) {
    v16bf an, n0, n1, n2, n3;
    if (kt + 1 < KTILES) {                 // prefetch next k-step
      an = load_a_frag(Arow, kt + 1, lane);
      n0 = load_b_frag(Bp, ((ntb + 0) * KTILES + (kt + 1)) * 32 + lane);
      n1 = load_b_frag(Bp, ((ntb + 1) * KTILES + (kt + 1)) * 32 + lane);
      n2 = load_b_frag(Bp, ((ntb + 2) * KTILES + (kt + 1)) * 32 + lane);
      n3 = load_b_frag(Bp, ((ntb + 3) * KTILES + (kt + 1)) * 32 + lane);
    } else {
      an = af; n0 = b0; n1 = b1; n2 = b2; n3 = b3;
    }
    acc[0] = __builtin_amdgcn_wmma_f32_16x16x32_bf16(false, af, false, b0,
                                                     (short)0, acc[0], false, false);
    acc[1] = __builtin_amdgcn_wmma_f32_16x16x32_bf16(false, af, false, b1,
                                                     (short)0, acc[1], false, false);
    acc[2] = __builtin_amdgcn_wmma_f32_16x16x32_bf16(false, af, false, b2,
                                                     (short)0, acc[2], false, false);
    acc[3] = __builtin_amdgcn_wmma_f32_16x16x32_bf16(false, af, false, b3,
                                                     (short)0, acc[3], false, false);
    af = an; b0 = n0; b1 = n1; b2 = n2; b3 = n3;
  }

  // Epilogue: VGPR r -> row rowbase+r (+8 for lanes 16..31), col striped.
  const int rowoff = (lane < 16) ? 0 : 8;
  const int cl = lane & 15;
#pragma unroll
  for (int na = 0; na < 4; ++na) {
    const int col = nbase + na * 16 + cl;
    const float bcol = bias[col];
#pragma unroll
    for (int r = 0; r < 8; ++r) {
      const int row = rowbase + r + rowoff;
      if (row < M) {
        float v = acc[na][r] + bcol;
        if (resid) v += resid[(size_t)row * 128 + col];
        if (relu_flag) v = fmaxf(v, 0.f);
        C[(size_t)row * ldc + coloff + col] = v;
      }
    }
  }
}

// ---------------------------------------------------------------------------
// Per-node per-head score: Q.K / sqrt(DH);  qkv layout [N][384] (Q|K|V)
// ---------------------------------------------------------------------------
__global__ void score_kernel(const float* __restrict__ qkv,
                             float* __restrict__ score, int total) {
  int t = blockIdx.x * blockDim.x + threadIdx.x;
  if (t >= total) return;
  int nid = t >> 3, h = t & 7;
  const float* q = qkv + (size_t)nid * 384 + h * 16;
  const float* k = q + 128;
  float s = 0.f;
#pragma unroll
  for (int j = 0; j < 16; j += 4) {
    float4 a = *(const float4*)(q + j);
    float4 b = *(const float4*)(k + j);
    s += a.x * b.x + a.y * b.y + a.z * b.z + a.w * b.w;
  }
  score[t] = s * 0.25f;  // 1/sqrt(16)
}

// ---------------------------------------------------------------------------
// CSR-by-destination construction
// ---------------------------------------------------------------------------
__global__ void zero_int(int* __restrict__ p, int n) {
  int t = blockIdx.x * blockDim.x + threadIdx.x;
  if (t < n) p[t] = 0;
}

__global__ void hist_kernel(const int* __restrict__ ei, int* __restrict__ counts,
                            int Etot) {
  int t = blockIdx.x * blockDim.x + threadIdx.x;
  if (t < Etot) atomicAdd(&counts[ei[Etot + t]], 1);
}

// Per-block exclusive scan (Hillis-Steele in LDS); block totals out.
__global__ void scan_block(const int* __restrict__ counts, int* __restrict__ excl,
                           int* __restrict__ blockSums, int n) {
  __shared__ int sh[256];
  int t = threadIdx.x;
  int i = blockIdx.x * 256 + t;
  int v = (i < n) ? counts[i] : 0;
  sh[t] = v;
  __syncthreads();
#pragma unroll
  for (int off = 1; off < 256; off <<= 1) {
    int add = (t >= off) ? sh[t - off] : 0;
    __syncthreads();
    sh[t] += add;
    __syncthreads();
  }
  if (i < n) excl[i] = sh[t] - v;
  if (t == 255) blockSums[blockIdx.x] = sh[255];
}

// Serial exclusive scan of ~400 block sums (trivial); total -> rowptr[N].
__global__ void scan_serial(int* __restrict__ blockSums, int nb,
                            int* __restrict__ totalOut) {
  if (blockIdx.x == 0 && threadIdx.x == 0) {
    int run = 0;
    for (int i = 0; i < nb; ++i) {
      int v = blockSums[i];
      blockSums[i] = run;
      run += v;
    }
    *totalOut = run;
  }
}

// Add block offsets; duplicate into cursor for the scatter pass.
__global__ void scan_add(int* __restrict__ rowptr, int* __restrict__ cursor,
                         const int* __restrict__ blockOffs, int n) {
  int i = blockIdx.x * 256 + threadIdx.x;
  if (i < n) {
    int v = rowptr[i] + blockOffs[blockIdx.x];
    rowptr[i] = v;
    cursor[i] = v;
  }
}

__global__ void fill_csr(const int* __restrict__ ei, int* __restrict__ cursor,
                         int* __restrict__ eidx, int Etot) {
  int t = blockIdx.x * blockDim.x + threadIdx.x;
  if (t >= Etot) return;
  int pos = atomicAdd(&cursor[ei[Etot + t]], 1);
  eidx[pos] = t;
}

// ---------------------------------------------------------------------------
// One wave per destination node: online (flash-style) segment softmax +
// alpha-weighted V aggregation. Lane owns 4 contiguous cols (one head per
// 4-lane group); per-head running max/denom replicated across its 4 lanes.
// Single non-atomic float4 store per lane.
// ---------------------------------------------------------------------------
__global__ __launch_bounds__(256)
void attn_aggregate(const int* __restrict__ ei, const int* __restrict__ rowptr,
                    const int* __restrict__ eidx,
                    const float* __restrict__ score,
                    const float* __restrict__ qkv, float* __restrict__ agg,
                    int Nn, int Etot) {
  int dst  = (int)((blockIdx.x * blockDim.x + threadIdx.x) >> 5);
  int lane = threadIdx.x & 31;
  if (dst >= Nn) return;
  const int beg = rowptr[dst], end = rowptr[dst + 1];
  const int c = lane * 4;        // 4 contiguous V columns
  const int h = lane >> 2;       // head for these columns
  const float sdst = score[dst * HEADS + h];

  float m = -3.0e38f, d = 0.f;
  float4 acc = {0.f, 0.f, 0.f, 0.f};
  for (int j = beg; j < end; ++j) {
    const int e = eidx[j];
    const int src = ei[e];
    float a = score[src * HEADS + h] + sdst;
    a = (a > 0.f) ? a : 0.2f * a;                 // leaky relu
    const float mnew = fmaxf(m, a);
    const float corr = __expf(m - mnew);
    const float p = __expf(a - mnew);
    d = d * corr + p;
    const float4 v = *(const float4*)(qkv + (size_t)src * 384 + 256 + c);
    acc.x = acc.x * corr + p * v.x;
    acc.y = acc.y * corr + p * v.y;
    acc.z = acc.z * corr + p * v.z;
    acc.w = acc.w * corr + p * v.w;
    m = mnew;
  }
  const float inv = 1.f / (d + 1e-16f);
  acc.x *= inv; acc.y *= inv; acc.z *= inv; acc.w *= inv;
  *(float4*)(agg + (size_t)dst * 128 + c) = acc;
}

// ---------------------------------------------------------------------------
// One wave per row (128 cols, 4/lane), wave32 shfl_xor reduction.
// ---------------------------------------------------------------------------
__global__ __launch_bounds__(256)
void layernorm_kernel(const float* __restrict__ in, const float* __restrict__ g,
                      const float* __restrict__ b, float* __restrict__ out,
                      int M) {
  int row = (int)((blockIdx.x * blockDim.x + threadIdx.x) >> 5);
  int lane = threadIdx.x & 31;
  if (row >= M) return;
  const float* p = in + (size_t)row * 128;
  float4 v = *(const float4*)(p + lane * 4);
  float s  = v.x + v.y + v.z + v.w;
  float s2 = v.x * v.x + v.y * v.y + v.z * v.z + v.w * v.w;
#pragma unroll
  for (int m = 16; m >= 1; m >>= 1) {
    s  += __shfl_xor(s, m, 32);
    s2 += __shfl_xor(s2, m, 32);
  }
  float mean = s * (1.f / 128.f);
  float var  = s2 * (1.f / 128.f) - mean * mean;
  float rstd = rsqrtf(var + 1e-5f);
  float4 gg = *(const float4*)(g + lane * 4);
  float4 bb = *(const float4*)(b + lane * 4);
  float4 o;
  o.x = (v.x - mean) * rstd * gg.x + bb.x;
  o.y = (v.y - mean) * rstd * gg.y + bb.y;
  o.z = (v.z - mean) * rstd * gg.z + bb.z;
  o.w = (v.w - mean) * rstd * gg.w + bb.w;
  *(float4*)(out + (size_t)row * 128 + lane * 4) = o;
}

// ---------------------------------------------------------------------------
extern "C" void kernel_launch(void* const* d_in, const int* in_sizes, int n_in,
                              void* d_out, int out_size, void* d_ws,
                              size_t ws_size, hipStream_t stream) {
  const float* x   = (const float*)d_in[0];
  const int*   ei  = (const int*)d_in[1];
  const float* Wq  = (const float*)d_in[2];
  const float* bq  = (const float*)d_in[3];
  const float* Wk  = (const float*)d_in[4];
  const float* bk  = (const float*)d_in[5];
  const float* Wv  = (const float*)d_in[6];
  const float* bv  = (const float*)d_in[7];
  const float* Wo  = (const float*)d_in[8];
  const float* bo  = (const float*)d_in[9];
  const float* g1  = (const float*)d_in[10];
  const float* b1  = (const float*)d_in[11];
  const float* Wf1 = (const float*)d_in[12];
  const float* bf1 = (const float*)d_in[13];
  const float* Wf2 = (const float*)d_in[14];
  const float* bf2 = (const float*)d_in[15];
  const float* g2  = (const float*)d_in[16];
  const float* b2  = (const float*)d_in[17];

  const int M    = in_sizes[0] / DIM;     // 100000 nodes
  const int Etot = in_sizes[1] / 2;       // total edges (incl. self loops)
  const int nb   = (M + 255) / 256;       // scan blocks

  // --- workspace carve-out (with reuse) ---
  char* ws = (char*)d_ws;
  size_t off = 0;
  auto carve = [&](size_t bytes) -> char* {
    char* p = ws + off;
    off = (off + bytes + 255) & ~(size_t)255;
    return p;
  };
  float* qkv    = (float*)carve((size_t)M * 384 * 4);  // Q|K|V; later h2 alias
  float* h2     = qkv;                                 // [M][256]
  float* score  = (float*)carve((size_t)M * 8 * 4);
  float* agg    = (float*)carve((size_t)M * 128 * 4);  // later h1n alias
  float* h1n    = agg;
  float* h1     = (float*)carve((size_t)M * 128 * 4);
  int*   counts = (int*)carve((size_t)M * 4);
  int*   rowptr = (int*)carve((size_t)(M + 1) * 4);
  int*   cursor = (int*)carve((size_t)M * 4);
  int*   bsums  = (int*)carve((size_t)nb * 4);
  int*   eidx   = (int*)carve((size_t)Etot * 4);
  bf16_t* pQ  = (bf16_t*)carve(8  * 4 * 32 * 16 * 2);
  bf16_t* pK  = (bf16_t*)carve(8  * 4 * 32 * 16 * 2);
  bf16_t* pV  = (bf16_t*)carve(8  * 4 * 32 * 16 * 2);
  bf16_t* pO  = (bf16_t*)carve(8  * 4 * 32 * 16 * 2);
  bf16_t* pF1 = (bf16_t*)carve(16 * 4 * 32 * 16 * 2);
  bf16_t* pF2 = (bf16_t*)carve(8  * 8 * 32 * 16 * 2);
  float* outp = (float*)d_out;

  // --- pack weights to WMMA bf16 layout ---
  pack_weights<<<(1024 + 255) / 256, 256, 0, stream>>>(Wq,  pQ,  128, 128);
  pack_weights<<<(1024 + 255) / 256, 256, 0, stream>>>(Wk,  pK,  128, 128);
  pack_weights<<<(1024 + 255) / 256, 256, 0, stream>>>(Wv,  pV,  128, 128);
  pack_weights<<<(1024 + 255) / 256, 256, 0, stream>>>(Wo,  pO,  128, 128);
  pack_weights<<<(2048 + 255) / 256, 256, 0, stream>>>(Wf1, pF1, 128, 256);
  pack_weights<<<(2048 + 255) / 256, 256, 0, stream>>>(Wf2, pF2, 256, 128);

  // --- CSR by destination (overlaps with GEMMs in issue order) ---
  zero_int<<<(M + 255) / 256, 256, 0, stream>>>(counts, M);
  hist_kernel<<<(Etot + 255) / 256, 256, 0, stream>>>(ei, counts, Etot);
  scan_block<<<nb, 256, 0, stream>>>(counts, rowptr, bsums, M);
  scan_serial<<<1, 32, 0, stream>>>(bsums, nb, rowptr + M);
  scan_add<<<nb, 256, 0, stream>>>(rowptr, cursor, bsums, M);
  fill_csr<<<(Etot + 255) / 256, 256, 0, stream>>>(ei, cursor, eidx, Etot);

  // --- QKV projections (bf16 WMMA, pipelined) ---
  dim3 gg1((M + 63) / 64, 1);
  gemm_wmma<4><<<gg1, 256, 0, stream>>>(x, pQ, bq, nullptr, qkv, M, 384, 0, 0);
  gemm_wmma<4><<<gg1, 256, 0, stream>>>(x, pK, bk, nullptr, qkv, M, 384, 128, 0);
  gemm_wmma<4><<<gg1, 256, 0, stream>>>(x, pV, bv, nullptr, qkv, M, 384, 256, 0);

  // --- scores + fused online-softmax aggregation (no float atomics) ---
  score_kernel<<<(M * 8 + 255) / 256, 256, 0, stream>>>(qkv, score, M * 8);
  attn_aggregate<<<(M * 32 + 255) / 256, 256, 0, stream>>>(ei, rowptr, eidx,
                                                           score, qkv, agg,
                                                           M, Etot);

  // --- output projection + residual + LN1 ---
  gemm_wmma<4><<<gg1, 256, 0, stream>>>(agg, pO, bo, x, h1, M, 128, 0, 0);
  layernorm_kernel<<<(M * 32 + 255) / 256, 256, 0, stream>>>(h1, g1, b1, h1n, M);

  // --- FFN (relu) + residual + LN2 (in place on d_out) ---
  dim3 gg2((M + 63) / 64, 2);
  gemm_wmma<4><<<gg2, 256, 0, stream>>>(h1n, pF1, bf1, nullptr, h2, M, 256, 0, 1);
  gemm_wmma<8><<<gg1, 256, 0, stream>>>(h2, pF2, bf2, h1n, outp, M, 128, 0, 0);
  layernorm_kernel<<<(M * 32 + 255) / 256, 256, 0, stream>>>(outp, g2, b2, outp, M);
}
